// DroneDockGAT_77472620085575
// MI455X (gfx1250) — matrix operand order repaired and checked by reading, not compiled, not run on hardware
//
#include <hip/hip_runtime.h>
#include <hip/hip_bf16.h>
#include <math.h>

// ---------------------------------------------------------------------------
// DroneDock GAT on MI455X (gfx1250): bf16 WMMA flash-style attention aggregate
// Round 3: ping-pong two-stage pipeline (no register-rotation moves).
// ---------------------------------------------------------------------------

typedef __attribute__((ext_vector_type(16))) __bf16 v16bf;
typedef __attribute__((ext_vector_type(8)))  float  v8f;

#define ND    8000
#define NK    2000
#define FD    32
#define FK    16
#define HDIM  64
#define HH    8
#define KPAD  2048          // padded dock count for WMMA B tiles (zero filled)
#define HPF   (HH * HDIM)   // 512
#define GAT_ALPHA 0.2f
#define GAT_NEG_INF (-9.0e15f)

// ---------------------------------------------------------------------------
// 1) Input projection + ELU:  out[n,j] = elu(x[n,:]@W[:,j] + b[j]), H=64 cols
// ---------------------------------------------------------------------------
__global__ void gat_proj_elu(const float* __restrict__ x,
                             const float* __restrict__ W,
                             const float* __restrict__ b,
                             float* __restrict__ out,
                             int Fin,
                             float* __restrict__ out2) {
  const int n = blockIdx.x;
  const int j = threadIdx.x;               // 64 threads
  __shared__ float xs[FD];
  if (j < Fin) xs[j] = x[n * Fin + j];
  __syncthreads();
  float acc = b[j];
  for (int d = 0; d < Fin; ++d) acc = fmaf(xs[d], W[d * HDIM + j], acc);
  float v = acc > 0.f ? acc : (__expf(acc) - 1.f);   // ELU
  out[n * HDIM + j] = v;
  if (out2) out2[n * HDIM + j] = v;
}

// ---------------------------------------------------------------------------
// 2) Fold attention vectors into Wa:  b1[h,d] = Wa[h,d,:]@a1[h], same for b2
// ---------------------------------------------------------------------------
__global__ void gat_prep_attn_vec(const float* __restrict__ Wa,
                                  const float* __restrict__ a1,
                                  const float* __restrict__ a2,
                                  float* __restrict__ b1,
                                  float* __restrict__ b2) {
  const int t = blockIdx.x * blockDim.x + threadIdx.x;  // 512 = HH*HDIM
  if (t >= HH * HDIM) return;
  const int h = t >> 6, d = t & 63;
  const float* w = Wa + (h * HDIM + d) * HDIM;
  float s1 = 0.f, s2 = 0.f;
  for (int f = 0; f < HDIM; ++f) {
    s1 = fmaf(w[f], a1[h * HDIM + f], s1);
    s2 = fmaf(w[f], a2[h * HDIM + f], s2);
  }
  b1[t] = s1; b2[t] = s2;
}

// ---------------------------------------------------------------------------
// 3) Per-head scores: out[h*N+n] = feat[n,:] . bvec[h,:]
// ---------------------------------------------------------------------------
__global__ void gat_rowdot(const float* __restrict__ feat,
                           const float* __restrict__ bvec,
                           float* __restrict__ out, int N) {
  const int t = blockIdx.x * blockDim.x + threadIdx.x;
  if (t >= HH * N) return;
  const int h = t / N, n = t % N;
  const float* f  = feat + n * HDIM;
  const float* bb = bvec + h * HDIM;
  float s = 0.f;
  for (int i = 0; i < HDIM; ++i) s = fmaf(f[i], bb[i], s);
  out[t] = s;
}

// ---------------------------------------------------------------------------
// 4) WhkT[h][f][k] = (h_dock[k,:] @ Wa[h,:,f]) in bf16, k padded to KPAD=2048
// ---------------------------------------------------------------------------
__global__ void gat_whk_prep(const float* __restrict__ hdock,
                             const float* __restrict__ Wa,
                             __bf16* __restrict__ whkT) {
  const int t = blockIdx.x * blockDim.x + threadIdx.x; // HH*HDIM*KPAD
  const int k  = t & (KPAD - 1);
  const int hf = t / KPAD;
  const int h = hf >> 6, f = hf & 63;
  float s = 0.f;
  if (k < NK) {
    const float* hd = hdock + k * HDIM;
    const float* w  = Wa + h * HDIM * HDIM + f;       // stride HDIM over d
    for (int d = 0; d < HDIM; ++d) s = fmaf(hd[d], w[d * HDIM], s);
  }
  whkT[t] = (__bf16)s;                                // zero pad for k>=NK
}

// ---------------------------------------------------------------------------
// 5) Pass 1: masked-softmax row stats. One wave32 per (head, drone) row.
// ---------------------------------------------------------------------------
__global__ void gat_softmax_stats(const int* __restrict__ adj,
                                  const float* __restrict__ sd,
                                  const float* __restrict__ sk,
                                  float* __restrict__ m_out,
                                  float* __restrict__ z_out) {
  const int lane = threadIdx.x & 31;
  const int wid  = blockIdx.x * (blockDim.x >> 5) + (threadIdx.x >> 5);
  const int h = wid / ND, n = wid % ND;
  const float sdv = sd[h * ND + n];
  const int*   arow = adj + (long)n * NK;
  const float* skh  = sk + h * NK;
  float m = -INFINITY, s = 0.f;
  for (int k = lane; k < NK; k += 32) {
    float x = sdv + skh[k];
    float e = (arow[k] > 0) ? (x > 0.f ? x : GAT_ALPHA * x) : GAT_NEG_INF;
    float nm = fmaxf(m, e);
    s = s * __expf(m - nm) + __expf(e - nm);
    m = nm;
  }
  for (int off = 16; off > 0; off >>= 1) {   // wave32 (m,s) combine
    float om = __shfl_xor(m, off, 32);
    float os = __shfl_xor(s, off, 32);
    float nm = fmaxf(m, om);
    s = s * __expf(m - nm) + os * __expf(om - nm);
    m = nm;
  }
  if (lane == 0) { m_out[h * ND + n] = m; z_out[h * ND + n] = s; }
}

// ---------------------------------------------------------------------------
// 6) Pass 2: WMMA aggregation with ping-pong software pipeline.
//    63 chunks of 32 k-values = 31 even/odd pairs + 1 tail. Each stage
//    consumes registers loaded one stage earlier and prefetches one ahead.
//    No buffer-rotation moves -> no WAR hazard v_movs/v_nops in the loop.
// ---------------------------------------------------------------------------
__global__ void __launch_bounds__(128)
gat_agg_wmma(const int* __restrict__ adj,
             const float* __restrict__ sd,
             const float* __restrict__ sk,
             const float* __restrict__ m_in,
             const float* __restrict__ z_in,
             const __bf16* __restrict__ whkT,
             float* __restrict__ hp) {
  const int lane = threadIdx.x & 31;
  const int wave = threadIdx.x >> 5;            // 0..3
  const int tile = blockIdx.x * 4 + wave;       // 0..499
  const int h    = blockIdx.y;                  // 0..7
  const int half = lane >> 4;
  const int l16  = lane & 15;
  const int n0   = tile * 16;
  const int n    = n0 + l16;                    // A-matrix row owned by lane

  const float sdv = sd[h * ND + n];
  const float mv  = m_in[h * ND + n];
  const int*    arow = adj + (long)n * NK;
  const float*  skh  = sk + h * NK;
  const __bf16* bbase = whkT + (long)(h * HDIM + l16) * KPAD;
  const long hoff = 16 * half;

  // ---- ping-pong buffers (never rotated, only ever written by loads) ----
  v16bf bA0, bA1, bA2, bA3;    // B tiles, buffer A
  v16bf bB0, bB1, bB2, bB3;    // B tiles, buffer B
  int4   aA0[2], aA1[2];       // adj raw, buffer A (per blk)
  float4 sA0[2], sA1[2];       // sk  raw, buffer A
  int4   aB0[2], aB1[2];       // adj raw, buffer B
  float4 sB0[2], sB1[2];       // sk  raw, buffer B

  auto loadB = [&](long kb, v16bf& o0, v16bf& o1, v16bf& o2, v16bf& o3) {
    const __bf16* p = bbase + kb + hoff;
    o0 = *(const v16bf*)(p + 0L * 16 * KPAD);
    o1 = *(const v16bf*)(p + 1L * 16 * KPAD);
    o2 = *(const v16bf*)(p + 2L * 16 * KPAD);
    o3 = *(const v16bf*)(p + 3L * 16 * KPAD);
  };
  auto loadA = [&](int kb, int4 (&A0)[2], int4 (&A1)[2],
                   float4 (&S0)[2], float4 (&S1)[2]) {
#pragma unroll
    for (int blk = 0; blk < 2; ++blk) {
      // guard is per half (half covers k 16*half..16*half+15); NK % 16 == 0, kb % 32 == 0
      if (kb + 16 * half + 16 <= NK) {
        const int kstart = kb + 16 * half + 8 * blk;   // 8 consecutive k
        A0[blk] = *(const int4*)(arow + kstart);
        A1[blk] = *(const int4*)(arow + kstart + 4);
        S0[blk] = *(const float4*)(skh + kstart);
        S1[blk] = *(const float4*)(skh + kstart + 4);
      } else {  // padded block: w pairs with zero-padded B rows -> no effect
        A0[blk] = make_int4(0, 0, 0, 0);
        A1[blk] = make_int4(0, 0, 0, 0);
        S0[blk] = make_float4(0.f, 0.f, 0.f, 0.f);
        S1[blk] = make_float4(0.f, 0.f, 0.f, 0.f);
      }
    }
  };
  auto buildA = [&](const int4 (&A0)[2], const int4 (&A1)[2],
                    const float4 (&S0)[2], const float4 (&S1)[2]) -> v16bf {
    v16bf a;
#pragma unroll
    for (int blk = 0; blk < 2; ++blk) {
      const int   am[8] = {A0[blk].x, A0[blk].y, A0[blk].z, A0[blk].w,
                           A1[blk].x, A1[blk].y, A1[blk].z, A1[blk].w};
      const float sv[8] = {S0[blk].x, S0[blk].y, S0[blk].z, S0[blk].w,
                           S1[blk].x, S1[blk].y, S1[blk].z, S1[blk].w};
#pragma unroll
      for (int q = 0; q < 8; ++q) {
        float x = sdv + sv[q];
        float e = (am[q] > 0) ? (x > 0.f ? x : GAT_ALPHA * x) : GAT_NEG_INF;
        a[blk * 8 + q] = (__bf16)__expf(e - mv);
      }
    }
    return a;
  };

  v8f c0 = {}, c1 = {}, c2 = {}, c3 = {};
  auto mmac = [&](const v16bf& a, const v16bf& x0, const v16bf& x1,
                  const v16bf& x2, const v16bf& x3) {
    c0 = __builtin_amdgcn_wmma_f32_16x16x32_bf16(false, a, false, x0, (short)0, c0, false, false);
    c1 = __builtin_amdgcn_wmma_f32_16x16x32_bf16(false, a, false, x1, (short)0, c1, false, false);
    c2 = __builtin_amdgcn_wmma_f32_16x16x32_bf16(false, a, false, x2, (short)0, c2, false, false);
    c3 = __builtin_amdgcn_wmma_f32_16x16x32_bf16(false, a, false, x3, (short)0, c3, false, false);
  };

  // prologue: chunk 0 -> buffer A
  loadB(0, bA0, bA1, bA2, bA3);
  loadA(0, aA0, aA1, sA0, sA1);

  // 31 pairs cover chunks 0..61; chunk 62 (kb=1984) lands in buffer A
  int kb = 0;
  for (int it = 0; it < 31; ++it, kb += 64) {
    // even stage: consume buffer A (chunk kb), prefetch chunk kb+32 -> B
    loadB((long)kb + 32, bB0, bB1, bB2, bB3);
    loadA(kb + 32, aB0, aB1, sB0, sB1);
    {
      v16bf a = buildA(aA0, aA1, sA0, sA1);
      mmac(a, bA0, bA1, bA2, bA3);
    }
    // odd stage: consume buffer B (chunk kb+32), prefetch chunk kb+64 -> A
    loadB((long)kb + 64, bA0, bA1, bA2, bA3);   // kb+64 <= 1984 (valid chunk)
    loadA(kb + 64, aA0, aA1, sA0, sA1);
    {
      v16bf a = buildA(aB0, aB1, sB0, sB1);
      mmac(a, bB0, bB1, bB2, bB3);
    }
  }
  // tail: chunk 62 already resident in buffer A
  {
    v16bf a = buildA(aA0, aA1, sA0, sA1);
    mmac(a, bA0, bA1, bA2, bA3);
  }

  // ---- epilogue: deferred 1/Z normalization, ELU, head-concat store ----
#pragma unroll
  for (int i = 0; i < 8; ++i) {
    const int row = i + 8 * half;                 // C/D layout: M = i + 8*half
    const int nn  = n0 + row;
    const float zinv = 1.0f / z_in[h * ND + nn];
    float* dst = hp + (long)nn * HPF + h * HDIM + l16;
    float v0 = c0[i] * zinv; dst[0]  = v0 > 0.f ? v0 : (__expf(v0) - 1.f);
    float v1 = c1[i] * zinv; dst[16] = v1 > 0.f ? v1 : (__expf(v1) - 1.f);
    float v2 = c2[i] * zinv; dst[32] = v2 > 0.f ? v2 : (__expf(v2) - 1.f);
    float v3 = c3[i] * zinv; dst[48] = v3 > 0.f ? v3 : (__expf(v3) - 1.f);
  }
}

// ---------------------------------------------------------------------------
// 7) Final fusion: out[n,j] = hp[n,:] @ Wf[:,j] + bf[j]  (fp32 for accuracy)
// ---------------------------------------------------------------------------
__global__ void gat_fuse_out(const float* __restrict__ hp,
                             const float* __restrict__ Wf,
                             const float* __restrict__ bfv,
                             float* __restrict__ out) {
  const int n = blockIdx.x;
  const int j = threadIdx.x;                    // 64 threads
  __shared__ float row[HPF];
  for (int i = j; i < HPF; i += 64) row[i] = hp[(long)n * HPF + i];
  __syncthreads();
  float acc = bfv[j];
  for (int c = 0; c < HPF; ++c) acc = fmaf(row[c], Wf[c * HDIM + j], acc);
  out[n * HDIM + j] = acc;
}

// ---------------------------------------------------------------------------
extern "C" void kernel_launch(void* const* d_in, const int* in_sizes, int n_in,
                              void* d_out, int out_size, void* d_ws, size_t ws_size,
                              hipStream_t stream) {
  const float* raw_drone = (const float*)d_in[0];
  const float* raw_dock  = (const float*)d_in[1];
  const int*   adj       = (const int*)d_in[2];
  const float* Wpd = (const float*)d_in[3];
  const float* bpd = (const float*)d_in[4];
  const float* Wpk = (const float*)d_in[5];
  const float* bpk = (const float*)d_in[6];
  const float* Wa  = (const float*)d_in[7];
  const float* a1  = (const float*)d_in[8];
  const float* a2  = (const float*)d_in[9];
  const float* Wf  = (const float*)d_in[10];
  const float* bfv = (const float*)d_in[11];

  float* out_drone = (float*)d_out;               // [8000,64]
  float* hdock_out = (float*)d_out + ND * HDIM;   // [2000,64]

  // workspace carve-up (~22 MB), 256B aligned slices
  char* ws = (char*)d_ws;
  size_t off = 0;
  auto carve = [&](size_t bytes) {
    void* p = ws + off;
    off = (off + bytes + 255) & ~(size_t)255;
    return p;
  };
  float*  h_drone = (float*)carve((size_t)ND * HDIM * 4);          // 2 MB
  float*  h_dock  = (float*)carve((size_t)NK * HDIM * 4);          // 0.5 MB
  float*  b1v     = (float*)carve((size_t)HH * HDIM * 4);
  float*  b2v     = (float*)carve((size_t)HH * HDIM * 4);
  float*  sdv     = (float*)carve((size_t)HH * ND * 4);            // 256 KB
  float*  skv     = (float*)carve((size_t)HH * NK * 4);            // 64 KB
  __bf16* whkT    = (__bf16*)carve((size_t)HH * HDIM * KPAD * 2);  // 2 MB
  float*  m_buf   = (float*)carve((size_t)HH * ND * 4);            // 256 KB
  float*  z_buf   = (float*)carve((size_t)HH * ND * 4);            // 256 KB
  float*  hp      = (float*)carve((size_t)ND * HPF * 4);           // 16 MB
  (void)ws_size; (void)in_sizes; (void)n_in; (void)out_size;

  // 1) projections (+ ELU); dock projection also writes second output
  gat_proj_elu<<<ND, 64, 0, stream>>>(raw_drone, Wpd, bpd, h_drone, FD, nullptr);
  gat_proj_elu<<<NK, 64, 0, stream>>>(raw_dock,  Wpk, bpk, h_dock,  FK, hdock_out);

  // 2) fold attention vectors: b1 = Wa @ a1, b2 = Wa @ a2 (per head)
  gat_prep_attn_vec<<<2, 256, 0, stream>>>(Wa, a1, a2, b1v, b2v);

  // 3) scores sd[h,n], sk[h,k]
  gat_rowdot<<<(HH * ND + 255) / 256, 256, 0, stream>>>(h_drone, b1v, sdv, ND);
  gat_rowdot<<<(HH * NK + 255) / 256, 256, 0, stream>>>(h_dock,  b2v, skv, NK);

  // 4) dock features per head, bf16, k-major + zero pad (WMMA B operand)
  gat_whk_prep<<<(HH * HDIM * KPAD) / 256, 256, 0, stream>>>(h_dock, Wa, whkT);

  // 5) masked-softmax stats (adj is 64MB -> L2 resident across passes)
  gat_softmax_stats<<<(HH * ND) / 8, 256, 0, stream>>>(adj, sdv, skv, m_buf, z_buf);

  // 6) the 16.4 GFLOP aggregation on bf16 WMMA: grid (500 tiles / 4 waves, 8 heads)
  gat_agg_wmma<<<dim3(500 / 4, HH), 128, 0, stream>>>(adj, sdv, skv, m_buf, z_buf, whkT, hp);

  // 7) final linear fusion
  gat_fuse_out<<<ND, 64, 0, stream>>>(hp, Wf, bfv, out_drone);
}
